// DeorphaNN_23742579212372
// MI455X (gfx1250) — matrix operand order repaired and checked by reading, not compiled, not run on hardware
//
#include <hip/hip_runtime.h>
#include <hip/hip_bf16.h>

#define NN   20000
#define EE   320000
#define FINC 128
#define H    10
#define C    32
#define HC   320            // H*C
#define BB   64
#define RALL (EE + NN)      // 340000, multiple of 16
#define NT_W 20             // 320 / 16 column tiles
#define KT_W 4              // 128 / 32 k tiles
#define NEG_SLOPE 0.2f
#define BN_EPS 1e-5f

typedef __attribute__((ext_vector_type(16))) __bf16 v16bf;
typedef __attribute__((ext_vector_type(8)))  float  v8f;

union BF16x16 { v16bf v; uint4 q[2]; unsigned short s[16]; };

__device__ __forceinline__ unsigned short f2bf(float f) {
  unsigned u = __float_as_uint(f);
  unsigned r = u + 0x7FFFu + ((u >> 16) & 1u);   // round-to-nearest-even
  return (unsigned short)(r >> 16);
}
// monotone float<->uint mapping for atomicMax-based segment max
__device__ __forceinline__ unsigned enc_f32(float f) {
  unsigned u = __float_as_uint(f);
  return (u & 0x80000000u) ? ~u : (u | 0x80000000u);
}
__device__ __forceinline__ float dec_f32(unsigned u) {
  unsigned b = (u & 0x80000000u) ? (u & 0x7FFFFFFFu) : ~u;
  return __uint_as_float(b);
}

// ---------------- K1: pack W[128,320] (f32) -> WMMA B fragments (bf16) ------
// Layout: P[kt][nt][lane][16] bf16; lane: n = nt*16 + (lane&15), kh = lane>>4,
// element e covers K = kt*32 + (e<8 ? kh*8 + e : 16 + kh*8 + e-8)  (B rows striped across lanes).
__global__ void pack_weights(const float* __restrict__ Wl,
                             const float* __restrict__ Wr,
                             const float* __restrict__ We,
                             unsigned short* __restrict__ Pl,
                             unsigned short* __restrict__ Pr,
                             unsigned short* __restrict__ Pe) {
  int id = blockIdx.x * blockDim.x + threadIdx.x;
  if (id >= 3 * KT_W * NT_W * 32) return;
  int w    = id / (KT_W * NT_W * 32);
  int r    = id % (KT_W * NT_W * 32);
  int kt   = r / (NT_W * 32);
  int r2   = r % (NT_W * 32);
  int nt   = r2 / 32;
  int lane = r2 % 32;
  const float* W = (w == 0) ? Wl : (w == 1) ? Wr : We;
  unsigned short* P = (w == 0) ? Pl : (w == 1) ? Pr : Pe;
  int n  = nt * 16 + (lane & 15);
  int kh = lane >> 4;
  unsigned short* dst = P + ((size_t)((kt * NT_W + nt) * 32 + lane)) * 16;
  for (int e = 0; e < 16; ++e) {
    int k = kt * 32 + ((e < 8) ? (kh * 8 + e) : (16 + kh * 8 + (e - 8)));
    dst[e] = f2bf(W[(size_t)k * HC + n]);
  }
}

// ---------------- K2a: segment sum of edge_attr over dst -------------------
__global__ void seg_attr_sum(const int* __restrict__ edge_index,
                             const float* __restrict__ edge_attr,
                             float* __restrict__ sum_attr,
                             float* __restrict__ cnt) {
  int idx = blockIdx.x * blockDim.x + threadIdx.x;     // E*128 = 40.96M < 2^31
  if (idx >= EE * FINC) return;
  int e = idx >> 7;
  int c = idx & 127;
  int dst = edge_index[EE + e];
  atomicAdd(&sum_attr[(size_t)dst * FINC + c], edge_attr[(size_t)e * FINC + c]);
  if (c == 0) atomicAdd(&cnt[dst], 1.0f);
}

// ---------------- K2b: loop_attr = sum / max(cnt,1) in place ---------------
__global__ void loop_attr_div(float* __restrict__ sum_attr,
                              const float* __restrict__ cnt) {
  int idx = blockIdx.x * blockDim.x + threadIdx.x;
  if (idx >= NN * FINC) return;
  float c = cnt[idx >> 7];
  sum_attr[idx] = sum_attr[idx] / fmaxf(c, 1.0f);
}

// ---------------- K3: fused BatchNorm + x_l/x_r GEMM (WMMA bf16) -----------
// 128 threads (4 waves) per 16-row tile; 40 column tiles split across waves.
__global__ __launch_bounds__(128)
void node_gemm(const float* __restrict__ x,
               const float* __restrict__ gamma, const float* __restrict__ beta,
               const float* __restrict__ mean,  const float* __restrict__ var,
               const unsigned short* __restrict__ Pl,
               const unsigned short* __restrict__ Pr,
               const float* __restrict__ bl, const float* __restrict__ br,
               float* __restrict__ xl, float* __restrict__ xr) {
  __shared__ __align__(16) unsigned short Atile[16 * FINC];
  int t = threadIdx.x;
  int row0 = blockIdx.x * 16;                       // N = 1250*16 exactly
  {
    int c = t;                                      // one channel per thread
    float sc = gamma[c] * rsqrtf(var[c] + BN_EPS);
    float sh = beta[c] - mean[c] * sc;
    for (int r = 0; r < 16; ++r)
      Atile[r * FINC + c] = f2bf(x[(size_t)(row0 + r) * FINC + c] * sc + sh);
  }
  __syncthreads();

  int wave = t >> 5, lane = t & 31;
  int m = lane & 15, half = lane >> 4;
  BF16x16 afr[KT_W];                                // A frags, reused per tile
  for (int kt = 0; kt < KT_W; ++kt) {
    const unsigned short* base = &Atile[m * FINC + kt * 32 + 8 * half];
    afr[kt].q[0] = *(const uint4*)base;             // K = 8h .. 8h+7
    afr[kt].q[1] = *(const uint4*)(base + 16);      // K = 16+8h .. 16+8h+7
  }
  for (int ti = 0; ti < 10; ++ti) {
    int tt = wave * 10 + ti;                        // 0..39
    const unsigned short* P = (tt < NT_W) ? Pl : Pr;
    const float* bias       = (tt < NT_W) ? bl : br;
    float* out              = (tt < NT_W) ? xl : xr;
    int nt = (tt < NT_W) ? tt : tt - NT_W;
    v8f acc = {};
    for (int kt = 0; kt < KT_W; ++kt) {
      BF16x16 bfr;
      const unsigned short* bp = P + ((size_t)((kt * NT_W + nt) * 32 + lane)) * 16;
      bfr.q[0] = *(const uint4*)bp;
      bfr.q[1] = *(const uint4*)(bp + 8);
      acc = __builtin_amdgcn_wmma_f32_16x16x32_bf16(false, afr[kt].v,
                                                    false, bfr.v,
                                                    (short)0, acc, false, false);
    }
    int n = nt * 16 + (lane & 15);
    float bv = bias[n];
    int rbase = row0 + 8 * half;                    // D: row = vgpr + 8*(lane/16)
    for (int r = 0; r < 8; ++r)
      out[(size_t)(rbase + r) * HC + n] = acc[r] + bv;
  }
}

// ---------------- K4: edge GEMM (e = ea@W_e) fused with z/score ------------
__global__ __launch_bounds__(128)
void edge_gemm_score(const float* __restrict__ edge_attr,
                     const float* __restrict__ loop_attr,
                     const unsigned short* __restrict__ Pe,
                     const int* __restrict__ edge_index,
                     const float* __restrict__ xl, const float* __restrict__ xr,
                     const float* __restrict__ att,
                     float* __restrict__ score) {
  __shared__ __align__(16) unsigned short Atile[16 * FINC];
  __shared__ float Etile[16][HC];
  __shared__ float Satt[HC];
  int t = threadIdx.x;
  int row0 = blockIdx.x * 16;                       // RALL = 21250*16 exactly
  for (int i = t; i < HC; i += 128) Satt[i] = att[i];
  {
    int c = t;
    for (int r = 0; r < 16; ++r) {
      int rg = row0 + r;
      float v = (rg < EE) ? edge_attr[(size_t)rg * FINC + c]
                          : loop_attr[(size_t)(rg - EE) * FINC + c];
      Atile[r * FINC + c] = f2bf(v);
    }
  }
  __syncthreads();

  int wave = t >> 5, lane = t & 31;
  int m = lane & 15, half = lane >> 4;
  BF16x16 afr[KT_W];
  for (int kt = 0; kt < KT_W; ++kt) {
    const unsigned short* base = &Atile[m * FINC + kt * 32 + 8 * half];
    afr[kt].q[0] = *(const uint4*)base;
    afr[kt].q[1] = *(const uint4*)(base + 16);
  }
  for (int ti = 0; ti < 5; ++ti) {
    int nt = wave * 5 + ti;                         // 0..19
    v8f acc = {};
    for (int kt = 0; kt < KT_W; ++kt) {
      BF16x16 bfr;
      const unsigned short* bp = Pe + ((size_t)((kt * NT_W + nt) * 32 + lane)) * 16;
      bfr.q[0] = *(const uint4*)bp;
      bfr.q[1] = *(const uint4*)(bp + 8);
      acc = __builtin_amdgcn_wmma_f32_16x16x32_bf16(false, afr[kt].v,
                                                    false, bfr.v,
                                                    (short)0, acc, false, false);
    }
    int n = nt * 16 + (lane & 15);
    for (int r = 0; r < 8; ++r)
      Etile[8 * half + r][n] = acc[r];
  }
  __syncthreads();

  // z = leaky(x_l[src] + x_r[dst] + e); score = <z[h], att[h]>
  for (int p = t; p < 16 * H; p += 128) {
    int row = p / H, h = p % H;
    int rg = row0 + row;
    int srcn = (rg < EE) ? edge_index[rg]      : (rg - EE);
    int dstn = (rg < EE) ? edge_index[EE + rg] : (rg - EE);
    const float* xls = xl + (size_t)srcn * HC + h * C;
    const float* xrd = xr + (size_t)dstn * HC + h * C;
    float s = 0.f;
    for (int c = 0; c < C; ++c) {
      float z = xls[c] + xrd[c] + Etile[row][h * C + c];
      z = (z > 0.f) ? z : NEG_SLOPE * z;
      s += z * Satt[h * C + c];
    }
    score[(size_t)rg * H + h] = s;
  }
}

// ---------------- K5: segment max of score over dst ------------------------
__global__ void seg_max(const int* __restrict__ edge_index,
                        const float* __restrict__ score,
                        unsigned* __restrict__ mseg) {
  int idx = blockIdx.x * blockDim.x + threadIdx.x;   // RALL*H = 3.4M
  if (idx >= RALL * H) return;
  int rg = idx / H, h = idx % H;
  int dstn = (rg < EE) ? edge_index[EE + rg] : (rg - EE);
  atomicMax(&mseg[dstn * H + h], enc_f32(score[idx]));
}

// ---------------- K6: alpha = exp(score-m); denom = segsum(alpha) ----------
__global__ void alpha_denom(const int* __restrict__ edge_index,
                            float* __restrict__ score,
                            const unsigned* __restrict__ mseg,
                            float* __restrict__ denom) {
  int idx = blockIdx.x * blockDim.x + threadIdx.x;
  if (idx >= RALL * H) return;
  int rg = idx / H, h = idx % H;
  int dstn = (rg < EE) ? edge_index[EE + rg] : (rg - EE);
  float mv = dec_f32(mseg[dstn * H + h]);
  float a = __expf(score[idx] - mv);
  score[idx] = a;                                     // reuse buffer for alpha
  atomicAdd(&denom[dstn * H + h], a);
}

// ---------------- K7: out[dst,h,:] += (alpha/denom) * x_l[src,h,:] ---------
__global__ void msg_accum(const int* __restrict__ edge_index,
                          const float* __restrict__ alpha,
                          const float* __restrict__ denom,
                          const float* __restrict__ xl,
                          float* __restrict__ acc) {
  int idx = blockIdx.x * blockDim.x + threadIdx.x;
  if (idx >= RALL * H) return;
  int rg = idx / H, h = idx % H;
  int srcn = (rg < EE) ? edge_index[rg]      : (rg - EE);
  int dstn = (rg < EE) ? edge_index[EE + rg] : (rg - EE);
  float a = alpha[idx] / denom[dstn * H + h];
  const float* xs = xl + (size_t)srcn * HC + h * C;
  float* ad = acc + (size_t)dstn * HC + h * C;
  for (int c = 0; c < C; ++c) atomicAdd(&ad[c], a * xs[c]);
}

// ---------------- K8: head mean + bias + relu + graph pooling --------------
__global__ void node_out_pool(const float* __restrict__ acc,
                              const float* __restrict__ bias_gat,
                              const int* __restrict__ batch,
                              float* __restrict__ gsum,
                              float* __restrict__ gcnt) {
  int idx = blockIdx.x * blockDim.x + threadIdx.x;   // N*C = 640k
  if (idx >= NN * C) return;
  int n = idx / C, c = idx % C;
  float s = 0.f;
  for (int h = 0; h < H; ++h) s += acc[(size_t)n * HC + h * C + c];
  s = s * (1.0f / H) + bias_gat[c];
  s = fmaxf(s, 0.f);
  int b = batch[n];
  atomicAdd(&gsum[b * C + c], s);
  if (c == 0) atomicAdd(&gcnt[b], 1.0f);
}

// ---------------- K9: final linear head ------------------------------------
__global__ void final_head(const float* __restrict__ gsum,
                           const float* __restrict__ gcnt,
                           const float* __restrict__ Wh,
                           const float* __restrict__ bh,
                           float* __restrict__ out) {
  int idx = threadIdx.x;
  if (idx >= BB * 2) return;
  int b = idx / 2, j = idx % 2;
  float invc = 1.0f / fmaxf(gcnt[b], 1.0f);
  float s = 0.f;
  for (int c = 0; c < C; ++c) s += (gsum[b * C + c] * invc) * Wh[c * 2 + j];
  out[idx] = s + bh[j];
}

extern "C" void kernel_launch(void* const* d_in, const int* in_sizes, int n_in,
                              void* d_out, int out_size, void* d_ws, size_t ws_size,
                              hipStream_t stream) {
  (void)in_sizes; (void)n_in; (void)out_size; (void)ws_size;
  const float* x          = (const float*)d_in[0];
  const int*   edge_index = (const int*)d_in[1];
  const float* edge_attr  = (const float*)d_in[2];
  const int*   batch      = (const int*)d_in[3];
  const float* bn_gamma   = (const float*)d_in[4];
  const float* bn_beta    = (const float*)d_in[5];
  const float* bn_mean    = (const float*)d_in[6];
  const float* bn_var     = (const float*)d_in[7];
  const float* W_l        = (const float*)d_in[8];
  const float* b_l        = (const float*)d_in[9];
  const float* W_r        = (const float*)d_in[10];
  const float* b_r        = (const float*)d_in[11];
  const float* W_e        = (const float*)d_in[12];
  const float* att        = (const float*)d_in[13];
  const float* bias_gat   = (const float*)d_in[14];
  const float* W_head     = (const float*)d_in[15];
  const float* b_head     = (const float*)d_in[16];
  float* out = (float*)d_out;

  // workspace partitioning (256B aligned)
  char* ws = (char*)d_ws;
  size_t off = 0;
  auto alloc = [&](size_t bytes) -> void* {
    void* p = ws + off;
    off += (bytes + 255) & ~(size_t)255;
    return p;
  };
  const size_t PACK_B = (size_t)KT_W * NT_W * 32 * 16 * sizeof(unsigned short);
  unsigned short* Pl = (unsigned short*)alloc(PACK_B);
  unsigned short* Pr = (unsigned short*)alloc(PACK_B);
  unsigned short* Pe = (unsigned short*)alloc(PACK_B);
  float* xl       = (float*)alloc((size_t)NN * HC * 4);
  float* xr       = (float*)alloc((size_t)NN * HC * 4);
  float* sum_attr = (float*)alloc((size_t)NN * FINC * 4);   // becomes loop_attr
  float* cnt      = (float*)alloc((size_t)NN * 4);
  float* score    = (float*)alloc((size_t)RALL * H * 4);    // becomes alpha
  unsigned* mseg  = (unsigned*)alloc((size_t)NN * H * 4);
  float* denom    = (float*)alloc((size_t)NN * H * 4);
  float* acc      = (float*)alloc((size_t)NN * HC * 4);
  float* gsum     = (float*)alloc((size_t)BB * C * 4);
  float* gcnt     = (float*)alloc((size_t)BB * 4);

  // zero-init accumulators (capture-safe async memsets)
  hipMemsetAsync(sum_attr, 0, (size_t)NN * FINC * 4, stream);
  hipMemsetAsync(cnt,      0, (size_t)NN * 4, stream);
  hipMemsetAsync(mseg,     0, (size_t)NN * H * 4, stream);  // 0 == -max in enc
  hipMemsetAsync(denom,    0, (size_t)NN * H * 4, stream);
  hipMemsetAsync(acc,      0, (size_t)NN * HC * 4, stream);
  hipMemsetAsync(gsum,     0, (size_t)BB * C * 4, stream);
  hipMemsetAsync(gcnt,     0, (size_t)BB * 4, stream);

  pack_weights<<<(3 * KT_W * NT_W * 32 + 127) / 128, 128, 0, stream>>>(
      W_l, W_r, W_e, Pl, Pr, Pe);

  seg_attr_sum<<<(EE * FINC + 255) / 256, 256, 0, stream>>>(
      edge_index, edge_attr, sum_attr, cnt);
  loop_attr_div<<<(NN * FINC + 255) / 256, 256, 0, stream>>>(sum_attr, cnt);

  node_gemm<<<NN / 16, 128, 0, stream>>>(
      x, bn_gamma, bn_beta, bn_mean, bn_var, Pl, Pr, b_l, b_r, xl, xr);

  edge_gemm_score<<<RALL / 16, 128, 0, stream>>>(
      edge_attr, sum_attr /*loop_attr*/, Pe, edge_index, xl, xr, att, score);

  int segN = RALL * H;
  seg_max<<<(segN + 255) / 256, 256, 0, stream>>>(edge_index, score, mseg);
  alpha_denom<<<(segN + 255) / 256, 256, 0, stream>>>(edge_index, score, mseg, denom);
  msg_accum<<<(segN + 255) / 256, 256, 0, stream>>>(edge_index, score, denom, xl, acc);

  node_out_pool<<<(NN * C + 255) / 256, 256, 0, stream>>>(
      acc, bias_gat, batch, gsum, gcnt);
  final_head<<<1, 128, 0, stream>>>(gsum, gcnt, W_head, b_head, out);
}
